// SemanticMatchingLoss_54795192762647
// MI455X (gfx1250) — matrix operand order repaired and checked by reading, not compiled, not run on hardware
//
#include <hip/hip_runtime.h>

// ---------------------------------------------------------------------------
// Problem sizes (fixed by the reference)
// ---------------------------------------------------------------------------
static constexpr int   BN = 8192;   // batch
static constexpr int   DK = 1024;   // feature dim
static constexpr int   CK = 256;    // concept dim
static constexpr float MARGIN    = 0.2f;
static constexpr float THRESHOLD = 0.5f;

static constexpr size_t IMG_ELEMS = (size_t)BN * DK;   // 8,388,608
static constexpr size_t CON_ELEMS = (size_t)BN * CK;   // 2,097,152

// Workspace layout (bytes)
static constexpr size_t OFF_ACC = 0;                                  // 2 floats
static constexpr size_t OFF_IMG = 256;                                // bf16 image  [BN,DK]
static constexpr size_t OFF_TXT = OFF_IMG + IMG_ELEMS * 2;            // bf16 text   [BN,DK]
static constexpr size_t OFF_CHI = OFF_TXT + IMG_ELEMS * 2;            // bf16 concept hi [BN,CK]
static constexpr size_t OFF_CLO = OFF_CHI + CON_ELEMS * 2;            // bf16 concept lo [BN,CK]

// Tiling: workgroup owns 32 rows; each wave computes 32x32 output patches
// (2x2 grid of 16x16 WMMA tiles). 8 waves split the 256 column-tile pairs,
// so each wave walks JJ_PER_WAVE = 32 column pairs per pass.
static constexpr int ROWS_PER_BLOCK = 32;
static constexpr int JT2_COUNT      = BN / 32;        // 256
static constexpr int NWAVES         = 8;
static constexpr int JJ_PER_WAVE    = JT2_COUNT / NWAVES;   // 32

// ---------------------------------------------------------------------------
// WMMA types / helpers (CDNA5 wave32)
// ---------------------------------------------------------------------------
typedef __attribute__((ext_vector_type(16))) __bf16 v16bf;
typedef __attribute__((ext_vector_type(8)))  float  v8f;

__device__ __forceinline__ v8f wmma_bf16(v16bf a, v16bf b, v8f c) {
    // D = A(16x32 bf16) * B(32x16 bf16) + C(16x16 f32)
    return __builtin_amdgcn_wmma_f32_16x16x32_bf16(
        /*neg_a=*/false, a, /*neg_b=*/false, b,
        /*c_mod=*/(short)0, c, /*reuse_a=*/false, /*reuse_b=*/false);
}

// A fragment: lane L -> row (L%16); VGPR0-3 = K[h*8 .. h*8+7], VGPR4-7 = K[16+h*8 ..]
__device__ __forceinline__ v16bf load_a_frag(const __bf16* rowPtr, int k0, int h) {
    v16bf r;
    const int4* p0 = reinterpret_cast<const int4*>(rowPtr + k0 + h * 8);
    const int4* p1 = reinterpret_cast<const int4*>(rowPtr + k0 + 16 + h * 8);
    reinterpret_cast<int4*>(&r)[0] = *p0;
    reinterpret_cast<int4*>(&r)[1] = *p1;
    return r;
}

// B fragment: lane L -> col (L%16); VGPR0-3 = K[h*8 .. h*8+7], VGPR4-7 = K[16+h*8 ..]
__device__ __forceinline__ v16bf load_b_frag(const __bf16* rowPtr, int k0, int h) {
    v16bf r;
    const int4* p0 = reinterpret_cast<const int4*>(rowPtr + k0 + h * 8);
    const int4* p1 = reinterpret_cast<const int4*>(rowPtr + k0 + 16 + h * 8);
    reinterpret_cast<int4*>(&r)[0] = *p0;
    reinterpret_cast<int4*>(&r)[1] = *p1;
    return r;
}

// Order-preserving float<->uint encoding so ds_max_rtn_u32 implements float max.
__device__ __forceinline__ unsigned f2ord(float f) {
    unsigned u = __float_as_uint(f);
    return (u & 0x80000000u) ? ~u : (u | 0x80000000u);
}
__device__ __forceinline__ float ord2f(unsigned u) {
    return (u & 0x80000000u) ? __uint_as_float(u & 0x7FFFFFFFu) : __uint_as_float(~u);
}

// 2x2 grid of 16x16 sim tiles (bf16, K=1024): 8 b128 loads feed 4 WMMAs per k-step.
__device__ __forceinline__ void sim_2x2(
    const __bf16* __restrict__ aRow0, const __bf16* __restrict__ aRow1,
    const __bf16* __restrict__ txt, int jBase, int mA, int hA, v8f s[2][2])
{
    const __bf16* bRow0 = txt + (size_t)(jBase + mA) * DK;
    const __bf16* bRow1 = bRow0 + (size_t)16 * DK;

    s[0][0] = (v8f){}; s[0][1] = (v8f){}; s[1][0] = (v8f){}; s[1][1] = (v8f){};
#pragma unroll 4
    for (int k0 = 0; k0 < DK; k0 += 32) {
        v16bf a0 = load_a_frag(aRow0, k0, hA);
        v16bf a1 = load_a_frag(aRow1, k0, hA);
        v16bf b0 = load_b_frag(bRow0, k0, hA);
        v16bf b1 = load_b_frag(bRow1, k0, hA);
        s[0][0] = wmma_bf16(a0, b0, s[0][0]);
        s[0][1] = wmma_bf16(a0, b1, s[0][1]);
        s[1][0] = wmma_bf16(a1, b0, s[1][0]);
        s[1][1] = wmma_bf16(a1, b1, s[1][1]);
    }
}

// 2x2 grid of 16x16 gram tiles (split-bf16, K=256): hi*hi + hi*lo + lo*hi
// recovers ~fp32 accuracy for the discrete 0.5 threshold.
__device__ __forceinline__ void gram_2x2(
    const __bf16* __restrict__ chRow0, const __bf16* __restrict__ chRow1,
    const __bf16* __restrict__ clRow0, const __bf16* __restrict__ clRow1,
    const __bf16* __restrict__ cHi, const __bf16* __restrict__ cLo,
    int jBase, int mA, int hA, v8f g[2][2])
{
    const __bf16* bhRow0 = cHi + (size_t)(jBase + mA) * CK;
    const __bf16* bhRow1 = bhRow0 + (size_t)16 * CK;
    const __bf16* blRow0 = cLo + (size_t)(jBase + mA) * CK;
    const __bf16* blRow1 = blRow0 + (size_t)16 * CK;

    g[0][0] = (v8f){}; g[0][1] = (v8f){}; g[1][0] = (v8f){}; g[1][1] = (v8f){};
#pragma unroll 2
    for (int k0 = 0; k0 < CK; k0 += 32) {
        v16bf ah0 = load_a_frag(chRow0, k0, hA);
        v16bf ah1 = load_a_frag(chRow1, k0, hA);
        v16bf al0 = load_a_frag(clRow0, k0, hA);
        v16bf al1 = load_a_frag(clRow1, k0, hA);
        v16bf bh0 = load_b_frag(bhRow0, k0, hA);
        v16bf bh1 = load_b_frag(bhRow1, k0, hA);
        v16bf bl0 = load_b_frag(blRow0, k0, hA);
        v16bf bl1 = load_b_frag(blRow1, k0, hA);
        g[0][0] = wmma_bf16(ah0, bh0, g[0][0]);
        g[0][1] = wmma_bf16(ah0, bh1, g[0][1]);
        g[1][0] = wmma_bf16(ah1, bh0, g[1][0]);
        g[1][1] = wmma_bf16(ah1, bh1, g[1][1]);
        g[0][0] = wmma_bf16(ah0, bl0, g[0][0]);
        g[0][1] = wmma_bf16(ah0, bl1, g[0][1]);
        g[1][0] = wmma_bf16(ah1, bl0, g[1][0]);
        g[1][1] = wmma_bf16(ah1, bl1, g[1][1]);
        g[0][0] = wmma_bf16(al0, bh0, g[0][0]);
        g[0][1] = wmma_bf16(al0, bh1, g[0][1]);
        g[1][0] = wmma_bf16(al1, bh0, g[1][0]);
        g[1][1] = wmma_bf16(al1, bh1, g[1][1]);
    }
}

// ---------------------------------------------------------------------------
// Kernel 0: fp32 -> bf16 conversion (+ hi/lo split for concepts) + accum init
// ---------------------------------------------------------------------------
__global__ void convert_init_kernel(const float* __restrict__ img,
                                    const float* __restrict__ txt,
                                    const float* __restrict__ con,
                                    __bf16* __restrict__ imgB,
                                    __bf16* __restrict__ txtB,
                                    __bf16* __restrict__ cHi,
                                    __bf16* __restrict__ cLo,
                                    float* __restrict__ accum)
{
    size_t i = (size_t)blockIdx.x * blockDim.x + threadIdx.x;
    if (i == 0) { accum[0] = 0.0f; accum[1] = 0.0f; }
    if (i < IMG_ELEMS) {
        imgB[i] = (__bf16)img[i];
        txtB[i] = (__bf16)txt[i];
    }
    if (i < CON_ELEMS) {
        float x  = con[i];
        __bf16 h = (__bf16)x;
        cHi[i] = h;
        cLo[i] = (__bf16)(x - (float)h);
    }
}

// ---------------------------------------------------------------------------
// Kernel 1: fused loss. One workgroup (8 waves) per 32-row block.
//   Pass A: sim + gram -> hard_neg / row flags; pos-mask bits (diag excluded)
//           cached in LDS (1 uint32 per lane per column pair).
//   Pass B: sim ONLY (gram never recomputed) + cached mask -> hinge sum/count.
// ---------------------------------------------------------------------------
__global__ void __launch_bounds__(256)
loss_main_kernel(const __bf16* __restrict__ img, const __bf16* __restrict__ txt,
                 const __bf16* __restrict__ cHi, const __bf16* __restrict__ cLo,
                 float* __restrict__ accum)
{
    // Cached positive-mask bits: maskLds[jj*256 + tid], bit = (mi*2+ni)*8 + r.
    // Consecutive tids -> consecutive dwords: conflict-free.
    __shared__ unsigned maskLds[JJ_PER_WAVE * 256];          // 32 KB
    __shared__ unsigned sHard[ROWS_PER_BLOCK];               // ordered-uint row max over negs
    __shared__ unsigned sPos[ROWS_PER_BLOCK];
    __shared__ unsigned sNeg[ROWS_PER_BLOCK];
    __shared__ float    sHn[ROWS_PER_BLOCK];
    __shared__ unsigned sValid[ROWS_PER_BLOCK];

    const int tid  = threadIdx.x;
    const int lane = tid & 31;
    const int wave = tid >> 5;
    const int mA   = lane & 15;    // A row within tile / B column within tile
    const int hA   = lane >> 4;    // K-half select (A/B), row-half select (C/D)
    const int mOff = hA * 8;       // C/D rows held by this lane: mOff..mOff+7
    const int rowBase = blockIdx.x * ROWS_PER_BLOCK;

    if (tid < ROWS_PER_BLOCK) { sHard[tid] = 0u; sPos[tid] = 0u; sNeg[tid] = 0u; }
    __syncthreads();

    // Per-wave A-row pointers (fixed across the column sweep)
    const __bf16* aRow0  = img + (size_t)(rowBase + mA) * DK;
    const __bf16* aRow1  = aRow0 + (size_t)16 * DK;
    const __bf16* chRow0 = cHi + (size_t)(rowBase + mA) * CK;
    const __bf16* chRow1 = chRow0 + (size_t)16 * CK;
    const __bf16* clRow0 = cLo + (size_t)(rowBase + mA) * CK;
    const __bf16* clRow1 = clRow0 + (size_t)16 * CK;

    // ---- Pass A: sim + gram -> hard negative, row flags, cached pos bits ----
    float hpart[2][8];
#pragma unroll
    for (int mi = 0; mi < 2; ++mi)
#pragma unroll
        for (int r = 0; r < 8; ++r) hpart[mi][r] = -3.402823466e38f;
    unsigned posAny = 0u, negAny = 0u;   // bit index = mi*8 + r

    for (int jj = 0; jj < JJ_PER_WAVE; ++jj) {   // uniform per wave -> EXEC all-1 at WMMAs
        const int jBase = (jj * NWAVES + wave) * 32;
        v8f s[2][2], g[2][2];
        sim_2x2(aRow0, aRow1, txt, jBase, mA, hA, s);
        gram_2x2(chRow0, chRow1, clRow0, clRow1, cHi, cLo, jBase, mA, hA, g);

        unsigned pmask = 0u;
#pragma unroll
        for (int mi = 0; mi < 2; ++mi) {
#pragma unroll
            for (int ni = 0; ni < 2; ++ni) {
#pragma unroll
                for (int r = 0; r < 8; ++r) {
                    const int gRow = rowBase + mi * 16 + mOff + r;
                    const int gCol = jBase + ni * 16 + mA;
                    const bool diag = (gRow == gCol);
                    const bool pos  = (g[mi][ni][r] > THRESHOLD) && !diag;
                    const bool neg  = !pos && !diag;
                    if (pos) {
                        pmask |= (1u << ((mi * 2 + ni) * 8 + r));
                        posAny |= (1u << (mi * 8 + r));
                    }
                    if (neg) {
                        negAny |= (1u << (mi * 8 + r));
                        hpart[mi][r] = fmaxf(hpart[mi][r], s[mi][ni][r]);
                    }
                }
            }
        }
        maskLds[jj * 256 + tid] = pmask;
    }

#pragma unroll
    for (int mi = 0; mi < 2; ++mi) {
#pragma unroll
        for (int r = 0; r < 8; ++r) {
            const int row = mi * 16 + mOff + r;
            if ((negAny >> (mi * 8 + r)) & 1u) {
                atomicMax(&sHard[row], f2ord(hpart[mi][r]));   // ds_max_rtn_u32
                atomicOr(&sNeg[row], 1u);
            }
            if ((posAny >> (mi * 8 + r)) & 1u) atomicOr(&sPos[row], 1u);
        }
    }
    __syncthreads();

    if (tid < ROWS_PER_BLOCK) {
        const bool v = (sPos[tid] != 0u) && (sNeg[tid] != 0u);
        sValid[tid] = v ? 1u : 0u;
        sHn[tid]    = v ? ord2f(sHard[tid]) : 0.0f;
    }
    __syncthreads();

    // ---- Pass B: sim only + cached pos bits -> hinge accumulation ----
    float lsum = 0.0f, lcnt = 0.0f;
    for (int jj = 0; jj < JJ_PER_WAVE; ++jj) {
        const int jBase = (jj * NWAVES + wave) * 32;
        v8f s[2][2];
        sim_2x2(aRow0, aRow1, txt, jBase, mA, hA, s);
        const unsigned pmask = maskLds[jj * 256 + tid];
#pragma unroll
        for (int mi = 0; mi < 2; ++mi) {
#pragma unroll
            for (int ni = 0; ni < 2; ++ni) {
#pragma unroll
                for (int r = 0; r < 8; ++r) {
                    const int row = mi * 16 + mOff + r;
                    // pos bit already excludes the diagonal
                    if (((pmask >> ((mi * 2 + ni) * 8 + r)) & 1u) && (sValid[row] != 0u)) {
                        const float h = MARGIN - s[mi][ni][r] + sHn[row];
                        lsum += (h > 0.0f) ? h : 0.0f;
                        lcnt += 1.0f;
                    }
                }
            }
        }
    }

    // wave32 reduction, then one global f32 atomic per wave
#pragma unroll
    for (int off = 16; off > 0; off >>= 1) {
        lsum += __shfl_xor(lsum, off, 32);
        lcnt += __shfl_xor(lcnt, off, 32);
    }
    if (lane == 0) {
        atomicAdd(&accum[0], lsum);
        atomicAdd(&accum[1], lcnt);
    }
}

// ---------------------------------------------------------------------------
// Kernel 2: finalize  loss = num_pos > 0 ? sum / num_pos : 0
// ---------------------------------------------------------------------------
__global__ void finalize_kernel(const float* __restrict__ accum, float* __restrict__ out)
{
    out[0] = (accum[1] > 0.0f) ? (accum[0] / accum[1]) : 0.0f;
}

// ---------------------------------------------------------------------------
// Launch
// ---------------------------------------------------------------------------
extern "C" void kernel_launch(void* const* d_in, const int* in_sizes, int n_in,
                              void* d_out, int out_size, void* d_ws, size_t ws_size,
                              hipStream_t stream)
{
    const float* img = (const float*)d_in[0];   // [8192,1024]
    const float* txt = (const float*)d_in[1];   // [8192,1024]
    const float* con = (const float*)d_in[2];   // [8192, 256]

    char* ws = (char*)d_ws;
    float*  accum = (float*)(ws + OFF_ACC);
    __bf16* imgB  = (__bf16*)(ws + OFF_IMG);
    __bf16* txtB  = (__bf16*)(ws + OFF_TXT);
    __bf16* cHi   = (__bf16*)(ws + OFF_CHI);
    __bf16* cLo   = (__bf16*)(ws + OFF_CLO);

    const int cvtBlocks = (int)((IMG_ELEMS + 255) / 256);   // 32768
    convert_init_kernel<<<cvtBlocks, 256, 0, stream>>>(img, txt, con, imgB, txtB, cHi, cLo, accum);

    loss_main_kernel<<<BN / ROWS_PER_BLOCK, 256, 0, stream>>>(imgB, txtB, cHi, cLo, accum);

    finalize_kernel<<<1, 1, 0, stream>>>(accum, (float*)d_out);
}
